// QuantumLLM_38826504356541
// MI455X (gfx1250) — matrix operand order repaired and checked by reading, not compiled, not run on hardware
//
#include <hip/hip_runtime.h>
#include <math.h>

#define D 256
#define S 256
#define BATCH 2
#define ROWS (BATCH * S)          // 512
#define PHI_F 1.6180339887498949f
#define EPS_F 1e-6f

typedef __attribute__((ext_vector_type(16))) _Float16 v16h;
typedef __attribute__((ext_vector_type(8)))  _Float16 v8h;
typedef __attribute__((ext_vector_type(8)))  float    v8f;

// ---------------- block reductions (wave32, blockDim == 256) ------------------
__device__ __forceinline__ float blk_sum(float v, float* lds) {
#pragma unroll
  for (int o = 16; o > 0; o >>= 1) v += __shfl_xor(v, o, 32);
  const int w = threadIdx.x >> 5;
  if ((threadIdx.x & 31) == 0) lds[w] = v;
  __syncthreads();
  if (threadIdx.x == 0) {
    float x = lds[0];
#pragma unroll
    for (int i = 1; i < 8; ++i) x += lds[i];
    lds[8] = x;
  }
  __syncthreads();
  float r = lds[8];
  __syncthreads();
  return r;
}

// Fused 4-way block sum: one barrier pass amortized over 4 reductions.
__device__ __forceinline__ void blk_sum4(float v[4], float* lds /*36 floats*/) {
#pragma unroll
  for (int u = 0; u < 4; ++u) {
#pragma unroll
    for (int o = 16; o > 0; o >>= 1) v[u] += __shfl_xor(v[u], o, 32);
  }
  const int w = threadIdx.x >> 5;
  if ((threadIdx.x & 31) == 0) {
#pragma unroll
    for (int u = 0; u < 4; ++u) lds[w * 4 + u] = v[u];
  }
  __syncthreads();
  if (threadIdx.x < 4) {
    float x = 0.f;
#pragma unroll
    for (int i = 0; i < 8; ++i) x += lds[i * 4 + threadIdx.x];
    lds[32 + threadIdx.x] = x;
  }
  __syncthreads();
#pragma unroll
  for (int u = 0; u < 4; ++u) v[u] = lds[32 + u];
  __syncthreads();
}

// ---------------- 1. embedding gather (f32 + f16 copies) ----------------------
__global__ void k_gather(const int* __restrict__ ids, const float* __restrict__ emb,
                         float* __restrict__ ground, _Float16* __restrict__ gh) {
  const int row = blockIdx.x;
  const int d = threadIdx.x;
  const float v = emb[ids[row] * D + d];
  ground[row * D + d] = v;
  gh[row * D + d] = (_Float16)v;
}

// ---------------- 1b. weight f32 -> f16 ---------------------------------------
__global__ void k_wconv(const float* __restrict__ w, _Float16* __restrict__ wh) {
  const int idx = blockIdx.x * blockDim.x + threadIdx.x;  // 3*D*D = 196608
  wh[idx] = (_Float16)w[idx];
}

// ---------------- 2. WMMA GEMM: exc[l] = ground @ W_l^T + b_l -----------------
// One wave computes a 16x32 output slab (two 16x16 WMMA tiles sharing the A
// fragment). K=256 fully unrolled -> 16 v_wmma per wave, f16 operands loaded
// directly as aligned 16B/32B vectors (no cvt in the hot loop).
// A (16x32 f16): lanes 0-15: M=lane, elems 0..15 -> K=k0..+15;
//                lanes 16-31 take K offset +16.
// B (32x16 f16): lane -> N=lane&15, 16 contiguous K (lanes>=16: K offset +16);
//                B[k][n] = W[n][k] so each lane reads a contiguous W row slice.
__global__ void k_wmma_gemm(const _Float16* __restrict__ gh,
                            const _Float16* __restrict__ wh,
                            const float* __restrict__ lt_b,
                            float* __restrict__ exc) {
  const int level = blockIdx.x >> 5;        // 3 levels x 32 blocks
  const int blk   = blockIdx.x & 31;
  const int wave  = threadIdx.x >> 5;
  const int lane  = threadIdx.x & 31;
  const int tile  = blk * 8 + wave;         // 0..255 (32 M-tiles x 8 N-pairs)
  const int tm    = (tile & 31) << 4;
  const int tnp   = (tile >> 5) << 5;       // 32-wide N slab
  const int hi    = lane >> 4;
  const int ln    = lane & 15;

  const _Float16* __restrict__ arow  = gh + (tm + ln) * D;
  const _Float16* __restrict__ brow0 = wh + level * D * D + (tnp + ln) * D;
  const _Float16* __restrict__ brow1 = brow0 + 16 * D;
  const float* __restrict__ bias     = lt_b + level * D;
  float* __restrict__ out            = exc + level * (ROWS * D);

  v8f c0 = {}, c1 = {};
#pragma unroll
  for (int k0 = 0; k0 < D; k0 += 32) {
    const int ka = k0 + hi * 16;
    const int kb = k0 + hi * 16;
    const v8h alo = *(const v8h*)(arow + ka);
    const v8h ahi = *(const v8h*)(arow + ka + 8);
    const v16h a = __builtin_shufflevector(alo, ahi, 0, 1, 2, 3, 4, 5, 6, 7,
                                           8, 9, 10, 11, 12, 13, 14, 15);
    const v16h b0 = *(const v16h*)(brow0 + kb);
    const v16h b1 = *(const v16h*)(brow1 + kb);
    c0 = __builtin_amdgcn_wmma_f32_16x16x32_f16(false, a, false, b0, (short)0,
                                                c0, false, false);
    c1 = __builtin_amdgcn_wmma_f32_16x16x32_f16(false, a, false, b1, (short)0,
                                                c1, false, false);
  }
  const int n0 = tnp + ln;
  const int n1 = n0 + 16;
  const float bv0 = bias[n0];
  const float bv1 = bias[n1];
#pragma unroll
  for (int r = 0; r < 8; ++r) {
    const int m = tm + hi * 8 + r;
    out[m * D + n0] = c0[r] + bv0;
    out[m * D + n1] = c1[r] + bv1;
  }
}

// ---------------- 3. per-row energies: e0=||ground||, e1=||exc_c0||, e2=||exc_c1||
__global__ void k_energy(const float* __restrict__ ground, const float* __restrict__ exc,
                         const float* __restrict__ pf_imag, float* __restrict__ evec) {
  __shared__ float lds[9];
  const int row = blockIdx.x, d = threadIdx.x;
  const float g = ground[row * D + d];
  float s = blk_sum(g * g, lds);
  if (d == 0) evec[0 * ROWS + row] = sqrtf(s);
  for (int l = 0; l < 2; ++l) {
    const float mag = expf(-pf_imag[l * D + d] * (float)(l + 1) * PHI_F);
    const float x = exc[l * ROWS * D + row * D + d] * mag;
    s = blk_sum(x * x, lds);
    if (d == 0) evec[(l + 1) * ROWS + row] = sqrtf(s);
  }
}

// ---------------- 4. global max of each energy vector (512 rows) --------------
__global__ void k_max(const float* __restrict__ evec, float* __restrict__ emax) {
  __shared__ float lds[9];
  const int t = threadIdx.x;
  for (int l = 0; l < 3; ++l) {
    float m = fmaxf(evec[l * ROWS + t], evec[l * ROWS + t + 256]);
#pragma unroll
    for (int o = 16; o > 0; o >>= 1) m = fmaxf(m, __shfl_xor(m, o, 32));
    if ((t & 31) == 0) lds[t >> 5] = m;
    __syncthreads();
    if (t == 0) {
      float x = lds[0];
#pragma unroll
      for (int i = 1; i < 8; ++i) x = fmaxf(x, lds[i]);
      emax[l] = x;
    }
    __syncthreads();
  }
}

// ---------------- 5. superposition + normalize + interference -----------------
__global__ void k_state(const float* __restrict__ ground, const float* __restrict__ exc,
                        const float* __restrict__ pf_real, const float* __restrict__ pf_imag,
                        const float* __restrict__ evec, const float* __restrict__ emax,
                        float* __restrict__ sre, float* __restrict__ sim) {
  __shared__ float lds[9];
  const int row = blockIdx.x, d = threadIdx.x;
  float re = ground[row * D + d], im = 0.f;
#pragma unroll
  for (int l = 0; l < 3; ++l) {
    const float e = evec[l * ROWS + row];
    const float prob = 1.f / (1.f + expf(-e / emax[l]));
    const float lp = (float)(l + 1) * PHI_F;
    const float ang = pf_real[l * D + d] * lp;
    const float mag = expf(-pf_imag[l * D + d] * lp);
    const float x = exc[l * ROWS * D + row * D + d] * mag;
    re += x * cosf(ang) * prob;
    im += x * sinf(ang) * prob;
  }
  const float n1 = sqrtf(blk_sum(re * re + im * im, lds));
  re /= n1; im /= n1;
  // interference: sum_j cos(p_i - p_j) = cos p_i * C + sin p_i * Sn
  const float ph = atan2f(im, re);
  const float cp = cosf(ph), sp = sinf(ph);
  const float C  = blk_sum(cp, lds);
  const float Sn = blk_sum(sp, lds);
  const float intf = cp * C + sp * Sn;
  const float orr = re * intf, oi = im * intf;
  const float n2 = sqrtf(blk_sum(orr * orr + oi * oi, lds));
  sre[row * D + d] = orr / n2;
  sim[row * D + d] = oi / n2;
}

// ---------------- 6. global phase: mean over sequence of angle ----------------
__global__ void k_gphase(const float* __restrict__ re, const float* __restrict__ im,
                         float* __restrict__ gph) {
  const int t = blockIdx.x * blockDim.x + threadIdx.x;  // 0..511 == b*D+d
  const int b = t >> 8, d = t & 255;
  float s = 0.f;
  for (int i = 0; i < S; ++i) {
    const int idx = (b * S + i) * D + d;
    s += atan2f(im[idx], re[idx]);
  }
  gph[t] = s * (1.f / (float)S);
}

// ---------------- 7. global quantum attention (fused, no [B,S,S,D] tensor) ---
// softmax arg = phase*D^-0.5 in [-pi/16, pi/16] -> max-subtraction elided.
// 4 keys per iteration: one fused reduction pass + 4 independent transcendental
// chains for the scheduler.
__global__ void k_attn(const float* __restrict__ inre, const float* __restrict__ inim,
                       const float* __restrict__ gph,
                       float* __restrict__ outre, float* __restrict__ outim) {
  __shared__ float lds[36];
  const int blk = blockIdx.x;
  const int b = blk >> 8, i = blk & 255, d = threadIdx.x;
  const int base = b * S * D;
  const float qr = inre[base + i * D + d];
  const float qi = inim[base + i * D + d];
  const float scale = 0.0625f;       // D^-0.5
  const float invS = 1.f / (float)S;
  float accr = 0.f, acci = 0.f;
  for (int j0 = 0; j0 < S; j0 += 4) {
    const int jn = (j0 + 4) & (S - 1);
    __builtin_prefetch(&inre[base + jn * D + d], 0, 0);
    __builtin_prefetch(&inim[base + jn * D + d], 0, 0);
    float kr[4], ki[4], ex[4], gi[4], sm[4];
#pragma unroll
    for (int u = 0; u < 4; ++u) {
      kr[u] = inre[base + (j0 + u) * D + d];
      ki[u] = inim[base + (j0 + u) * D + d];
    }
#pragma unroll
    for (int u = 0; u < 4; ++u) {
      const float dot = qr * kr[u] + qi * ki[u];
      const float cross = qi * kr[u] - qr * ki[u];
      float pd = cross / (fabsf(dot) + EPS_F);
      pd = pd * (1.f - 0.28125f * pd * pd);
      gi[u] = cosf(pd) * invS;
      const float ph = atan2f(qi - ki[u], qr - kr[u]);
      ex[u] = expf(ph * scale);
      sm[u] = ex[u];
    }
    blk_sum4(sm, lds);
#pragma unroll
    for (int u = 0; u < 4; ++u) {
      const float w = (ex[u] / sm[u]) * gi[u];
      accr += w * kr[u];
      acci += w * ki[u];
    }
  }
  const float g = gph[b * D + d];
  const float cg = cosf(g), sg = sinf(g);
  outre[base + i * D + d] = accr * cg - acci * sg;
  outim[base + i * D + d] = accr * sg + acci * cg;
}

// ---------------- 8. interleave re/im into [B,S,D,2] --------------------------
__global__ void k_pack(const float* __restrict__ re, const float* __restrict__ im,
                       float* __restrict__ out) {
  const int idx = blockIdx.x * blockDim.x + threadIdx.x;  // 0..131071
  out[idx * 2 + 0] = re[idx];
  out[idx * 2 + 1] = im[idx];
}

extern "C" void kernel_launch(void* const* d_in, const int* in_sizes, int n_in,
                              void* d_out, int out_size, void* d_ws, size_t ws_size,
                              hipStream_t stream) {
  (void)in_sizes; (void)n_in; (void)out_size; (void)ws_size;
  const int*   ids  = (const int*)d_in[0];
  const float* emb  = (const float*)d_in[1];
  const float* lt_w = (const float*)d_in[2];
  const float* lt_b = (const float*)d_in[3];
  const float* pfr  = (const float*)d_in[4];
  const float* pfi  = (const float*)d_in[5];
  float* out = (float*)d_out;
  float* ws  = (float*)d_ws;

  float* ground = ws;               // 131072 floats
  float* exc    = ws + 131072;      // 3 * 131072
  float* evec   = ws + 524288;      // 3 * 512
  float* emax   = ws + 525824;      // 3 (+pad)
  float* sre    = ws + 526336;      // 131072
  float* sim    = ws + 657408;      // 131072
  float* tre    = ws + 788480;      // 131072
  float* tim    = ws + 919552;      // 131072
  float* gph    = ws + 1050624;     // 512
  _Float16* gh  = (_Float16*)(ws + 1051136);   // 131072 halfs
  _Float16* wh  = (_Float16*)(ws + 1116672);   // 196608 halfs

  k_gather   <<<ROWS, D,   0, stream>>>(ids, emb, ground, gh);
  k_wconv    <<<768,  256, 0, stream>>>(lt_w, wh);
  k_wmma_gemm<<<96,   256, 0, stream>>>(gh, wh, lt_b, exc);
  k_energy   <<<ROWS, D,   0, stream>>>(ground, exc, pfi, evec);
  k_max      <<<1,    256, 0, stream>>>(evec, emax);
  k_state    <<<ROWS, D,   0, stream>>>(ground, exc, pfr, pfi, evec, emax, sre, sim);
  k_gphase   <<<2,    256, 0, stream>>>(sre, sim, gph);
  k_attn     <<<ROWS, 256, 0, stream>>>(sre, sim, gph, tre, tim);
  k_gphase   <<<2,    256, 0, stream>>>(tre, tim, gph);
  k_attn     <<<ROWS, 256, 0, stream>>>(tre, tim, gph, sre, sim);
  k_pack     <<<ROWS, 256, 0, stream>>>(sre, sim, out);
}
